// GRURnn_70377334112753
// MI455X (gfx1250) — matrix operand order repaired and checked by reading, not compiled, not run on hardware
//
#include <hip/hip_runtime.h>

// Bidirectional GRU (B=128, T=512, E=H=256) for gfx1250 / MI455X.
// Latency-bound serial recurrence -> one workgroup per (direction, 16-row
// batch tile), WMMA bf16 16x16x32 with f32 accumulate for both the fused
// input projection and the hidden projection each step.
// B-fragments are streamed from L2 each step (pointer laundered through an
// empty asm to stop LICM; cast to address_space(1) so they lower to
// global_load_b128 on LOADcnt only, decoupled from the LDS/DScnt pipe).

#define B_    128
#define T_    512
#define E_    256
#define H_    256
#define G_    768     // 3*H (gates r,z,n)
#define NCLS_ 10

#define FRAG_DW 256                       // 32 lanes * 8 dwords per bf16 B-fragment
#define NTILES  48                        // G/16 N-tiles
#define KSTEPS  8                         // 256/32 k-steps
#define NFRAGS_PER_MAT (NTILES * KSTEPS)  // 384 fragments per weight matrix
#define PW_DW   (2 * 2 * NFRAGS_PER_MAT * FRAG_DW)  // dirs * {Wih,Whh}
#define PW_BYTES ((size_t)PW_DW * 4)                 // 1.5 MB packed weights

#define AS1 __attribute__((address_space(1)))

typedef __attribute__((ext_vector_type(16))) __bf16        v16bf;
typedef __attribute__((ext_vector_type(8)))  float         v8f;
typedef __attribute__((ext_vector_type(4)))  unsigned int  u32x4;

static __device__ inline unsigned short f2bf(float f) {
  unsigned int u = __float_as_uint(f);
  unsigned int r = u + 0x7FFFu + ((u >> 16) & 1u);   // round-to-nearest-even
  return (unsigned short)(r >> 16);
}

static __device__ inline v8f splat8(float s) {
  v8f r = {s, s, s, s, s, s, s, s};
  return r;
}

// A-fragment (16x32 bf16, M x K) from an LDS tile stored row-major [16][256].
// Per ISA: lanes 0-15 -> M=lane, K groups {0..7,16..23}; lanes 16-31 -> M=lane-16,
// K groups {8..15,24..31}; dword j holds the K pair (k, k+1).  The two 4-dword
// groups are contiguous 16B chunks -> vectorizes to two ds_load_b128.
static __device__ inline v16bf load_a_frag(const unsigned short* lds, int lane, int kbase) {
  const int m    = lane & 15;
  const int half = lane >> 4;
  const unsigned short* row = lds + m * 256;
  union { unsigned int u[8]; v16bf v; } r;
#pragma unroll
  for (int j = 0; j < 8; ++j) {
    int k = kbase + ((j < 4) ? (half * 8 + 2 * j) : (16 + half * 8 + 2 * (j - 4)));
    r.u[j] = *(const unsigned int*)(row + k);
  }
  return r.v;
}

// B-fragment: 8 consecutive dwords per lane at base + idx (global memory).
static __device__ inline v16bf load_b_frag(const AS1 unsigned int* base, int idx) {
  union { u32x4 q[2]; v16bf v; } r;
  const AS1 u32x4* p = (const AS1 u32x4*)(base + idx);
  r.q[0] = p[0];
  r.q[1] = p[1];
  return r.v;
}

// ---------------------------------------------------------------------------
// Kernel A: pack W_ih/W_hh (f32, torch layout [3H][K]) into per-lane bf16
// B-fragments.  fid = ((dir*2 + which)*48 + ntile)*8 + kk, 256 dwords each.
// Within a fragment: lane L (group = L>>4, n = L&15), dword j -> element
// B[k, n] = W[ntile*16 + n][kk*32 + (j<4 ? group*8 + 2j : 16 + group*8 + 2(j-4)) (+1)].
// ---------------------------------------------------------------------------
__global__ __launch_bounds__(256) void pack_weights_kernel(
    const float* __restrict__ Wihf, const float* __restrict__ Whhf,
    const float* __restrict__ Wihb, const float* __restrict__ Whhb,
    unsigned int* __restrict__ pw) {
  const int fid = blockIdx.x;           // 0..1535
  const int kk    = fid & 7;
  const int r1    = fid >> 3;
  const int ntile = r1 % NTILES;
  const int r2    = r1 / NTILES;
  const int which = r2 & 1;             // 0 = W_ih (x), 1 = W_hh (h)
  const int dir   = r2 >> 1;            // 0 = fwd, 1 = bwd

  const float* W = dir ? (which ? Whhb : Wihb) : (which ? Whhf : Wihf);

  const int t    = threadIdx.x;         // 0..255 -> (lane, dword)
  const int lane = t >> 3;
  const int j    = t & 7;
  const int group = lane >> 4;
  const int nn    = lane & 15;
  const int K = kk * 32 + ((j < 4) ? (group * 8 + 2 * j) : (16 + group * 8 + 2 * (j - 4)));
  const int g = ntile * 16 + nn;        // weight row (gate output index)

  const float v0 = W[g * 256 + K];
  const float v1 = W[g * 256 + K + 1];
  const unsigned int d = (unsigned int)f2bf(v0) | ((unsigned int)f2bf(v1) << 16);
  pw[(size_t)fid * FRAG_DW + lane * 8 + j] = d;
}

// ---------------------------------------------------------------------------
// Kernel B: one workgroup = (direction, 16-row batch tile).  512 threads =
// 16 wave32; wave w owns hidden tile w (16 hidden units) -> gate N-tiles
// {w, w+16, w+32}.  Per step: fused x-projection GEMM + h-projection GEMM
// (each K=256 -> 8 WMMA k-steps), then GRU gate math, h kept in LDS.
// ---------------------------------------------------------------------------
__global__ __launch_bounds__(512, 1) void gru_dir_kernel(
    const int*   __restrict__ inputs,     // [128][512]
    const float* __restrict__ emb,        // [30000][256]
    const float* __restrict__ b_ih_f, const float* __restrict__ b_hh_f,
    const float* __restrict__ b_ih_b, const float* __restrict__ b_hh_b,
    const unsigned int* __restrict__ pw,  // packed weights
    float* __restrict__ hidden) {         // [128][512] (fwd | bwd)
  __shared__ unsigned short xA[16 * 256]; // x_t tile, bf16
  __shared__ unsigned short hA[16 * 256]; // h tile, bf16 (GEMM operand)
  __shared__ float          hF[16 * 256]; // h tile, f32 (state)

  const int tid   = threadIdx.x;
  const int lane  = tid & 31;
  const int wv    = tid >> 5;             // wave id 0..15 == hidden tile
  const int btile = blockIdx.x;           // 0..7
  const int dir   = blockIdx.y;           // 0..1
  const int bbase = btile * 16;

  const float* b_ih = dir ? b_ih_b : b_ih_f;
  const float* b_hh = dir ? b_hh_b : b_hh_f;
  const unsigned int* pwx = pw + (size_t)(dir * 2 + 0) * NFRAGS_PER_MAT * FRAG_DW;
  const unsigned int* pwh = pw + (size_t)(dir * 2 + 1) * NFRAGS_PER_MAT * FRAG_DW;

  const int nn   = lane & 15;
  const int half = lane >> 4;
  const int hid  = wv * 16 + nn;          // this lane's hidden column

  // Per-lane biases (loop-invariant); folded into accumulator init.
  const float bxr = b_ih[hid], bxz = b_ih[H_ + hid], bxn = b_ih[2 * H_ + hid];
  const float bhr = b_hh[hid], bhz = b_hh[H_ + hid], bhn = b_hh[2 * H_ + hid];

  // Fragment dword indices: idx = ((g*16 + wv)*KSTEPS + kk)*FRAG_DW + lane*8
  const int voff = lane * 8;              // per-lane part (VGPR offset)
  const int wvoff = wv * KSTEPS * FRAG_DW;

  // h0 = 0
  for (int i = tid; i < 16 * 256; i += 512) { hF[i] = 0.0f; hA[i] = 0; }

  const int m_ld   = tid >> 5;            // coop-load row 0..15
  const int col_ld = (tid & 31) * 8;      // coop-load col base

  for (int t = 0; t < T_; ++t) {
    const int tc = dir ? (T_ - 1 - t) : t;

    // --- embedding gather for x_t (16 rows x 256 cols, f32 -> bf16 LDS) ---
    {
      const int tok = inputs[(bbase + m_ld) * T_ + tc];
      const float* er = emb + (size_t)tok * E_ + col_ld;
#pragma unroll
      for (int i = 0; i < 8; ++i) xA[m_ld * 256 + col_ld + i] = f2bf(er[i]);
    }
    __syncthreads();   // xA ready; previous step's hA writes visible

    // Launder the packed-weight pointers so the B-fragment loads are NOT
    // loop-invariant (no hoist -> no scratch spills), then cast to global
    // address space so they lower to global_load_b128 (LOADcnt only).
    const unsigned int* pwx_t = pwx;
    const unsigned int* pwh_t = pwh;
    asm volatile("" : "+s"(pwx_t), "+s"(pwh_t));
    const AS1 unsigned int* gpwx = (const AS1 unsigned int*)pwx_t;
    const AS1 unsigned int* gpwh = (const AS1 unsigned int*)pwh_t;

    // --- GEMMs: accx = x_t @ W_ih^T + b_ih ; acch = h @ W_hh^T + b_hh ---
    v8f accx[3] = { splat8(bxr), splat8(bxz), splat8(bxn) };
    v8f acch[3] = { splat8(bhr), splat8(bhz), splat8(bhn) };

#pragma unroll
    for (int kk = 0; kk < KSTEPS; ++kk) {
      const v16bf a_x = load_a_frag(xA, lane, kk * 32);
      const v16bf a_h = load_a_frag(hA, lane, kk * 32);
#pragma unroll
      for (int g = 0; g < 3; ++g) {
        const int idx = wvoff + (g * 16 * KSTEPS + kk) * FRAG_DW + voff;
        const v16bf bxf = load_b_frag(gpwx, idx);
        accx[g] = __builtin_amdgcn_wmma_f32_16x16x32_bf16(
            false, a_x, false, bxf, (short)0, accx[g], false, false);
        const v16bf bhf = load_b_frag(gpwh, idx);
        acch[g] = __builtin_amdgcn_wmma_f32_16x16x32_bf16(
            false, a_h, false, bhf, (short)0, acch[g], false, false);
      }
    }

    // --- GRU gate math (C/D layout: lanes<16 -> M=v, lanes>=16 -> M=v+8) ---
    float hnew[8];
#pragma unroll
    for (int v = 0; v < 8; ++v) {
      const int brow = v + half * 8;
      const float r  = 1.0f / (1.0f + __expf(-(accx[0][v] + acch[0][v])));
      const float z  = 1.0f / (1.0f + __expf(-(accx[1][v] + acch[1][v])));
      const float nv = tanhf(accx[2][v] + r * acch[2][v]);
      const float ho = hF[brow * 256 + hid];   // owner-lane read, race-free
      hnew[v] = (1.0f - z) * nv + z * ho;
    }
    __syncthreads();   // all waves done reading hA before overwrite

#pragma unroll
    for (int v = 0; v < 8; ++v) {
      const int brow = v + half * 8;
      hF[brow * 256 + hid] = hnew[v];
      hA[brow * 256 + hid] = f2bf(hnew[v]);
    }
  }
  __syncthreads();

  // --- final hidden state -> workspace [128][512], fwd in [0,256) bwd in [256,512) ---
#pragma unroll
  for (int i = 0; i < 8; ++i) {
    hidden[(size_t)(bbase + m_ld) * (2 * H_) + dir * H_ + col_ld + i] =
        hF[m_ld * 256 + col_ld + i];
  }
}

// ---------------------------------------------------------------------------
// Kernel C: logits = hidden @ W_lin^T + b_lin   (128 x 10, K=512; trivial)
// ---------------------------------------------------------------------------
__global__ __launch_bounds__(256) void head_kernel(
    const float* __restrict__ hidden, const float* __restrict__ Wlin,
    const float* __restrict__ blin, float* __restrict__ out) {
  const int gid = blockIdx.x * blockDim.x + threadIdx.x;
  if (gid >= B_ * NCLS_) return;
  const int b = gid / NCLS_;
  const int c = gid % NCLS_;
  const float* hrow = hidden + (size_t)b * (2 * H_);
  const float* wrow = Wlin + (size_t)c * (2 * H_);
  float acc = blin[c];
#pragma unroll 4
  for (int j = 0; j < 2 * H_; ++j) acc += hrow[j] * wrow[j];
  out[gid] = acc;
}

extern "C" void kernel_launch(void* const* d_in, const int* in_sizes, int n_in,
                              void* d_out, int out_size, void* d_ws, size_t ws_size,
                              hipStream_t stream) {
  (void)in_sizes; (void)n_in; (void)out_size; (void)ws_size;

  const int*   inputs = (const int*)  d_in[0];
  const float* emb    = (const float*)d_in[1];
  const float* Wihf   = (const float*)d_in[2];
  const float* Whhf   = (const float*)d_in[3];
  const float* bihf   = (const float*)d_in[4];
  const float* bhhf   = (const float*)d_in[5];
  const float* Wihb   = (const float*)d_in[6];
  const float* Whhb   = (const float*)d_in[7];
  const float* bihb   = (const float*)d_in[8];
  const float* bhhb   = (const float*)d_in[9];
  const float* Wlin   = (const float*)d_in[10];
  const float* blin   = (const float*)d_in[11];
  float* out = (float*)d_out;

  unsigned int* pw = (unsigned int*)d_ws;
  float* hidden = (float*)((char*)d_ws + PW_BYTES);

  // A: pack weights to bf16 WMMA B-fragments (1536 fragments).
  pack_weights_kernel<<<dim3(2 * 2 * NTILES * KSTEPS), dim3(256), 0, stream>>>(
      Wihf, Whhf, Wihb, Whhb, pw);

  // B: 16 independent workgroups (8 batch tiles x 2 directions), full T loop each.
  gru_dir_kernel<<<dim3(8, 2), dim3(512), 0, stream>>>(
      inputs, emb, bihf, bhhf, bihb, bhhb, pw, hidden);

  // C: classifier head.
  head_kernel<<<dim3((B_ * NCLS_ + 255) / 256), dim3(256), 0, stream>>>(
      hidden, Wlin, blin, out);
}
